// BilinearAttention_85847806313080
// MI455X (gfx1250) — matrix operand order repaired and checked by reading, not compiled, not run on hardware
//
#include <hip/hip_runtime.h>

// ---------------------------------------------------------------------------
// BilinearAttention for MI455X (gfx1250, wave32, WMMA 16x16x32 bf16)
// ---------------------------------------------------------------------------
#define B_   2
#define T_   2048
#define C_   1024
#define H_   16
#define D_   64
#define M_   (B_ * T_)      // 4096 rows

typedef __attribute__((ext_vector_type(16))) __bf16 v16bf;
typedef __attribute__((ext_vector_type(8)))  float  v8f;

// ---- WMMA fragment helpers (layouts per CDNA5 ISA §7.12.2) ------

// A-matrix 16x32 bf16: lane&15 = row M; lane<16 -> K {0..7,16..23}; lane>=16 -> K {8..15,24..31}
// rowptr points at this lane's row, already offset to the 32-wide K chunk.
__device__ inline v16bf make_a_frag(const __bf16* rowptr, int hi) {
  v16bf a;
#pragma unroll
  for (int e = 0; e < 8; ++e) a[e] = rowptr[hi * 8 + e];
#pragma unroll
  for (int e = 0; e < 8; ++e) a[8 + e] = rowptr[16 + hi * 8 + e];
  return a;
}

// B-matrix 32x16 bf16 where the source is stored with B's column n as a
// contiguous row (i.e. source row n holds B[k][n] for k = 0..31).
// rowptr points at that row, already offset to the 32-wide K chunk.
__device__ inline v16bf make_b_frag_row(const __bf16* rowptr, int hi) {
  v16bf b;
#pragma unroll
  for (int e = 0; e < 8; ++e) b[e] = rowptr[hi * 8 + e];
#pragma unroll
  for (int e = 0; e < 8; ++e) b[8 + e] = rowptr[16 + hi * 8 + e];
  return b;
}

__device__ inline v8f wmma_bf16(v16bf a, v16bf b, v8f c) {
  return __builtin_amdgcn_wmma_f32_16x16x32_bf16(false, a, false, b, (short)0, c,
                                                 false, false);
}

// ---------------------------------------------------------------------------
// Kernel 1: projection  Out[b,h,t,d] = (x @ W^T), optional fused RoPE.
// Grid: (M_/64, H_). Block: 128 threads (4 waves); wave w owns rows w*16..w*16+15.
// ---------------------------------------------------------------------------
template <bool ROPE>
__global__ __launch_bounds__(128) void proj_kernel(const float* __restrict__ X,
                                                   const float* __restrict__ W,
                                                   __bf16* __restrict__ Out) {
  __shared__ __bf16 Xs[64][40];
  __shared__ __bf16 Ws[64][40];

  const int m0   = blockIdx.x * 64;
  const int h    = blockIdx.y;           // head index == N tile (64 wide)
  const int tid  = threadIdx.x;
  const int wave = tid >> 5;
  const int lane = tid & 31;
  const int l    = lane & 15;
  const int hi   = lane >> 4;

  v8f acc[4] = {};

  for (int k0 = 0; k0 < C_; k0 += 32) {
#pragma unroll
    for (int i = 0; i < 16; ++i) {
      int idx = tid + i * 128;           // 0..2047
      int r = idx >> 5, c = idx & 31;
      Xs[r][c] = (__bf16)X[(size_t)(m0 + r) * C_ + k0 + c];
      Ws[r][c] = (__bf16)W[(size_t)(h * 64 + r) * C_ + k0 + c];
    }
    __syncthreads();

    v16bf a = make_a_frag(&Xs[wave * 16 + l][0], hi);
#pragma unroll
    for (int j = 0; j < 4; ++j) {
      v16bf b = make_b_frag_row(&Ws[j * 16 + l][0], hi);  // B[k][n]=W[n][k]
      acc[j] = wmma_bf16(a, b, acc[j]);
    }
    __syncthreads();
  }

  // Store: element (m = m0 + wave*16 + hi*8 + r, d = j*16 + l)
  const int mbase = m0 + wave * 16 + hi * 8;
  if (ROPE) {
#pragma unroll
    for (int j = 0; j < 2; ++j) {
      int d = j * 16 + l;                               // 0..31 (first half)
      float inv_freq = __powf(10000.0f, -(float)(2 * d) / (float)D_);
#pragma unroll
      for (int r = 0; r < 8; ++r) {
        int m = mbase + r;
        int bb = m >> 11, t = m & (T_ - 1);
        float ang = (float)t * inv_freq;
        float cv = (float)(__bf16)__cosf(ang);          // bf16-rounded like ref
        float sv = (float)(__bf16)__sinf(ang);
        float x1 = acc[j][r], x2 = acc[j + 2][r];
        float y1 = x1 * cv + x2 * sv;
        float y2 = -x1 * sv + x2 * cv;
        size_t base = ((size_t)(bb * H_ + h) * T_ + t) * D_;
        Out[base + d]      = (__bf16)y1;
        Out[base + d + 32] = (__bf16)y2;
      }
    }
  } else {
#pragma unroll
    for (int j = 0; j < 4; ++j) {
      int d = j * 16 + l;
#pragma unroll
      for (int r = 0; r < 8; ++r) {
        int m = mbase + r;
        int bb = m >> 11, t = m & (T_ - 1);
        Out[((size_t)(bb * H_ + h) * T_ + t) * D_ + d] = (__bf16)acc[j][r];
      }
    }
  }
}

// ---------------------------------------------------------------------------
// Kernel 2: causal bilinear attention, no softmax.
//   y[q,:] = sum_{k<=q} (s1[q,k]/D)*(s2[q,k]/D) * v[k,:]
// Grid: (T_/64 query tiles, B_*H_). Block 128 (4 waves).
// V is stored transposed in LDS so every WMMA B-fragment (K1, K2, V) is a
// contiguous 32B row read -> vectorized ds_load_b128 pairs.
// ---------------------------------------------------------------------------
__global__ __launch_bounds__(128) void attn_kernel(
    const __bf16* __restrict__ Q1, const __bf16* __restrict__ K1,
    const __bf16* __restrict__ Q2, const __bf16* __restrict__ K2,
    const __bf16* __restrict__ V, __bf16* __restrict__ Y) {
  __shared__ __bf16 K1s[64][72];
  __shared__ __bf16 K2s[64][72];
  __shared__ __bf16 Vt[64][72];   // transposed: Vt[d][key]
  __shared__ __bf16 Ps[64][72];   // per-wave 16-row strips of P (bf16)

  const int qt   = blockIdx.x;
  const int bh   = blockIdx.y;
  const int tid  = threadIdx.x;
  const int wave = tid >> 5;
  const int lane = tid & 31;
  const int l    = lane & 15;
  const int hi   = lane >> 4;

  const size_t hd = (size_t)bh * T_ * D_;
  const __bf16* q1p = Q1 + hd;
  const __bf16* k1p = K1 + hd;
  const __bf16* q2p = Q2 + hd;
  const __bf16* k2p = K2 + hd;
  const __bf16* vp  = V + hd;

  // Q fragments for this wave's 16 query rows, kept in VGPRs across kv loop.
  const int qrow = qt * 64 + wave * 16 + l;
  v16bf q1f[2], q2f[2];
#pragma unroll
  for (int kc = 0; kc < 2; ++kc) {
    q1f[kc] = make_a_frag(q1p + (size_t)qrow * D_ + kc * 32, hi);
    q2f[kc] = make_a_frag(q2p + (size_t)qrow * D_ + kc * 32, hi);
  }

  v8f yacc[4] = {};
  const float inv_d = 1.0f / (float)D_;

  for (int kt = 0; kt <= qt; ++kt) {
    // Cooperative load of K1/K2 tiles (16B chunks) and V transposed.
#pragma unroll
    for (int i = 0; i < 4; ++i) {
      int cidx = tid + i * 128;          // 0..511
      int r = cidx >> 3, c8 = (cidx & 7) * 8;
      size_t goff = (size_t)(kt * 64 + r) * D_ + c8;
      *(uint4*)&K1s[r][c8] = *(const uint4*)&k1p[goff];
      *(uint4*)&K2s[r][c8] = *(const uint4*)&k2p[goff];
      uint4 vv = *(const uint4*)&vp[goff];
      const __bf16* ve = (const __bf16*)&vv;
#pragma unroll
      for (int e = 0; e < 8; ++e) Vt[c8 + e][r] = ve[e];   // transpose
    }
    __syncthreads();

    // S1 = Q1*K1^T, S2 = Q2*K2^T over D=64 (two 32-wide K chunks).
    v8f s1[4] = {}, s2[4] = {};
#pragma unroll
    for (int kc = 0; kc < 2; ++kc) {
#pragma unroll
      for (int j = 0; j < 4; ++j) {
        v16bf b1 = make_b_frag_row(&K1s[j * 16 + l][kc * 32], hi);
        s1[j] = wmma_bf16(q1f[kc], b1, s1[j]);
        v16bf b2 = make_b_frag_row(&K2s[j * 16 + l][kc * 32], hi);
        s2[j] = wmma_bf16(q2f[kc], b2, s2[j]);
      }
    }

    // Mask + scale + elementwise product -> bf16 P strip in LDS.
    const int qg0 = qt * 64 + wave * 16 + hi * 8;
#pragma unroll
    for (int j = 0; j < 4; ++j) {
      int kg = kt * 64 + j * 16 + l;
#pragma unroll
      for (int r = 0; r < 8; ++r) {
        int qg = qg0 + r;
        float p = (kg <= qg) ? (s1[j][r] * inv_d) * (s2[j][r] * inv_d) : 0.0f;
        Ps[wave * 16 + hi * 8 + r][j * 16 + l] = (__bf16)p;
      }
    }
    // Same-wave LDS store->load is kept in order by DScnt; no barrier needed
    // since each wave reads only its own 16-row strip.

    // Y += P * V (contract over the 64 keys of this tile).
    // B[k=key][n=d] = V[key][d] = Vt[d][key] -> contiguous row read.
#pragma unroll
    for (int kc = 0; kc < 2; ++kc) {
      v16bf pa = make_a_frag(&Ps[wave * 16 + l][kc * 32], hi);
#pragma unroll
      for (int j = 0; j < 4; ++j) {
        v16bf bv = make_b_frag_row(&Vt[j * 16 + l][kc * 32], hi);
        yacc[j] = wmma_bf16(pa, bv, yacc[j]);
      }
    }
    __syncthreads();   // protect K/V/P tiles before next iteration's loads
  }

  // Store y in [B, T, C] layout (bf16) for the output projection.
  const int bb = bh >> 4, h = bh & 15;
#pragma unroll
  for (int j = 0; j < 4; ++j) {
#pragma unroll
    for (int r = 0; r < 8; ++r) {
      int t = qt * 64 + wave * 16 + hi * 8 + r;
      Y[(size_t)(bb * T_ + t) * C_ + h * 64 + j * 16 + l] = (__bf16)yacc[j][r];
    }
  }
}

// ---------------------------------------------------------------------------
// Kernel 3: output projection  out = y @ Wout^T  (f32 result)
// Grid: (M_/64, C_/64). Block 128 (4 waves).
// ---------------------------------------------------------------------------
__global__ __launch_bounds__(128) void out_gemm_kernel(
    const __bf16* __restrict__ Yin, const float* __restrict__ W,
    float* __restrict__ Out) {
  __shared__ __bf16 Xs[64][40];
  __shared__ __bf16 Ws[64][40];

  const int m0   = blockIdx.x * 64;
  const int n0   = blockIdx.y * 64;
  const int tid  = threadIdx.x;
  const int wave = tid >> 5;
  const int lane = tid & 31;
  const int l    = lane & 15;
  const int hi   = lane >> 4;

  v8f acc[4] = {};

  for (int k0 = 0; k0 < C_; k0 += 32) {
#pragma unroll
    for (int i = 0; i < 16; ++i) {
      int idx = tid + i * 128;
      int r = idx >> 5, c = idx & 31;
      Xs[r][c] = Yin[(size_t)(m0 + r) * C_ + k0 + c];
      Ws[r][c] = (__bf16)W[(size_t)(n0 + r) * C_ + k0 + c];
    }
    __syncthreads();

    v16bf a = make_a_frag(&Xs[wave * 16 + l][0], hi);
#pragma unroll
    for (int j = 0; j < 4; ++j) {
      v16bf b = make_b_frag_row(&Ws[j * 16 + l][0], hi);
      acc[j] = wmma_bf16(a, b, acc[j]);
    }
    __syncthreads();
  }

  const int mbase = m0 + wave * 16 + hi * 8;
#pragma unroll
  for (int j = 0; j < 4; ++j) {
#pragma unroll
    for (int r = 0; r < 8; ++r) {
      Out[(size_t)(mbase + r) * C_ + n0 + j * 16 + l] = acc[j][r];
    }
  }
}

// ---------------------------------------------------------------------------
// Host-side launcher.
// Inputs: x, Wq1, Wk1, Wq2, Wk2, Wv, Wout (all f32). Output: f32 [B,T,C].
// Workspace: q1,k1,q2,k2,v (bf16 [B,H,T,D]) + y (bf16 [B,T,C]) = 48 MB.
// ---------------------------------------------------------------------------
extern "C" void kernel_launch(void* const* d_in, const int* in_sizes, int n_in,
                              void* d_out, int out_size, void* d_ws,
                              size_t ws_size, hipStream_t stream) {
  (void)in_sizes; (void)n_in; (void)out_size; (void)ws_size;
  const float* x    = (const float*)d_in[0];
  const float* Wq1  = (const float*)d_in[1];
  const float* Wk1  = (const float*)d_in[2];
  const float* Wq2  = (const float*)d_in[3];
  const float* Wk2  = (const float*)d_in[4];
  const float* Wv   = (const float*)d_in[5];
  const float* Wout = (const float*)d_in[6];
  float* out = (float*)d_out;

  __bf16* ws = (__bf16*)d_ws;
  const size_t per = (size_t)B_ * H_ * T_ * D_;   // 4M elems
  __bf16* q1 = ws + 0 * per;
  __bf16* k1 = ws + 1 * per;
  __bf16* q2 = ws + 2 * per;
  __bf16* k2 = ws + 3 * per;
  __bf16* v  = ws + 4 * per;
  __bf16* y  = ws + 5 * per;

  dim3 blk(128);
  dim3 gproj(M_ / 64, H_);
  proj_kernel<true><<<gproj, blk, 0, stream>>>(x, Wq1, q1);
  proj_kernel<true><<<gproj, blk, 0, stream>>>(x, Wk1, k1);
  proj_kernel<true><<<gproj, blk, 0, stream>>>(x, Wq2, q2);
  proj_kernel<true><<<gproj, blk, 0, stream>>>(x, Wk2, k2);
  proj_kernel<false><<<gproj, blk, 0, stream>>>(x, Wv, v);

  attn_kernel<<<dim3(T_ / 64, B_ * H_), blk, 0, stream>>>(q1, k1, q2, k2, v, y);

  out_gemm_kernel<<<dim3(M_ / 64, C_ / 64), blk, 0, stream>>>(y, Wout, out);
}
